// STICKYLlamaAttention_6717328850981
// MI455X (gfx1250) — matrix operand order repaired and checked by reading, not compiled, not run on hardware
//
#include <hip/hip_runtime.h>

// ---------------------------------------------------------------------------
// Llama attention layer for MI455X (gfx1250, wave32, WMMA + TDM async tensor).
// B=1, S=2048, HID=4096, NH=32, NKV=8, HD=128, GROUPS=4.
// GEMMs: v_wmma_f32_16x16x32_bf16 fed from LDS tiles staged by
// tensor_load_to_lds (double buffered, TENSORcnt-tracked).
// ---------------------------------------------------------------------------

typedef __attribute__((ext_vector_type(16))) __bf16 bf16x16;
typedef __attribute__((ext_vector_type(8)))  __bf16 bf16x8;
typedef __attribute__((ext_vector_type(8)))  float  f32x8;
typedef __attribute__((ext_vector_type(4)))  float  f32x4;
typedef __attribute__((ext_vector_type(4)))  unsigned int u32x4;
typedef __attribute__((ext_vector_type(8)))  int    i32x8;
typedef __attribute__((ext_vector_type(4)))  int    i32x4;

#define DEVFN static __device__ __forceinline__

static constexpr int S_LEN = 2048;
static constexpr int HID   = 4096;
static constexpr int NH    = 32;
static constexpr int NKV   = 8;
static constexpr int HD    = 128;

DEVFN f32x8 wmma_bf16(bf16x16 a, bf16x16 b, f32x8 c) {
  return __builtin_amdgcn_wmma_f32_16x16x32_bf16(
      /*neg_a=*/false, a, /*neg_b=*/false, b,
      /*c_mod=*/(short)0, c, /*reuse_a=*/false, /*reuse_b=*/false);
}

// LDS flat address: low 32 bits are the wave-relative LDS byte offset
// (ISA 10.2 aperture mapping).
DEVFN unsigned lds_addr_of(const void* p) {
  return (unsigned)(unsigned long long)p;
}

// ------------------------- TDM 2D tile load --------------------------------
// D# per CDNA5 ISA.  2D tile: tile_dim0 = cols
// (contiguous, data_size units), tile_dim1 = rows, row stride = stride_elems.
// elem_code: 1 = 2 bytes, 2 = 4 bytes.
DEVFN void tdm_load_2d(unsigned lds_byte_addr, const void* gptr,
                       unsigned tensor_d0, unsigned tensor_d1,
                       unsigned tile_cols, unsigned tile_rows,
                       unsigned stride_elems, unsigned elem_code) {
  unsigned long long ga = (unsigned long long)gptr;
  u32x4 g0;
  g0[0] = 1u;                                            // count=1, user mode
  g0[1] = lds_byte_addr;                                 // lds_addr [63:32]
  g0[2] = (unsigned)(ga & 0xFFFFFFFFu);                  // global_addr[31:0]
  g0[3] = ((unsigned)(ga >> 32) & 0x01FFFFFFu) | (2u << 30); // ga[56:32], type=2
  i32x8 g1;
  g1[0] = (int)(elem_code << 16);                        // mask=0, data_size
  g1[1] = (int)((tensor_d0 & 0xFFFFu) << 16);            // tensor_dim0[15:0]
  g1[2] = (int)((tensor_d0 >> 16) | ((tensor_d1 & 0xFFFFu) << 16));
  g1[3] = (int)((tensor_d1 >> 16) | (tile_cols << 16));  // tile_dim0
  g1[4] = (int)(tile_rows & 0xFFFFu);                    // tile_dim1, tile_dim2=0
  g1[5] = (int)stride_elems;                             // dim0_stride[31:0]
  g1[6] = 0;                                             // dim0_stride[47:32], dim1_stride lo
  g1[7] = 0;
  i32x4 z4 = {0, 0, 0, 0};
  i32x8 z8 = {0, 0, 0, 0, 0, 0, 0, 0};
  __builtin_amdgcn_tensor_load_to_lds(g0, g1, z4, z4, z8, 0);
}

// --------------------------- fragment loaders ------------------------------
// A fragment (16x32, MxK): lane L -> row (L&15); lanes 0-15 hold K {0..7,16..23},
// lanes 16-31 hold K {8..15,24..31}.   (ISA 7.12.2, 16-bit A 16x32)
DEVFN bf16x16 load_a(const __bf16* A, long lda, int row0, int k0, int lane) {
  const __bf16* p = A + (long)(row0 + (lane & 15)) * lda + (k0 + ((lane >> 4) << 3));
  bf16x8 lo = *(const bf16x8*)p;
  bf16x8 hi = *(const bf16x8*)(p + 16);
  bf16x16 f;
#pragma unroll
  for (int i = 0; i < 8; ++i) { f[i] = lo[i]; f[i + 8] = hi[i]; }
  return f;
}

DEVFN bf16x16 load_a(const float* A, long lda, int row0, int k0, int lane) {
  const float* p = A + (long)(row0 + (lane & 15)) * lda + (k0 + ((lane >> 4) << 3));
  f32x4 a0 = *(const f32x4*)p;
  f32x4 a1 = *(const f32x4*)(p + 4);
  f32x4 b0 = *(const f32x4*)(p + 16);
  f32x4 b1 = *(const f32x4*)(p + 20);
  bf16x16 f;
#pragma unroll
  for (int i = 0; i < 4; ++i) {
    f[i]      = (__bf16)a0[i];
    f[i + 4]  = (__bf16)a1[i];
    f[i + 8]  = (__bf16)b0[i];
    f[i + 12] = (__bf16)b1[i];
  }
  return f;
}

// B fragment (32x16, KxN), source stored row-major [N][K] (K contiguous):
// lane L -> col (L&15); lanes 0-15 hold K {0..7,16..23}, lanes 16-31 hold K {8..15,24..31}.
DEVFN bf16x16 load_b(const __bf16* B, long ldb, int n0, int k0, int lane) {
  const __bf16* p = B + (long)(n0 + (lane & 15)) * ldb + (k0 + ((lane >> 4) << 3));
  bf16x8 lo = *(const bf16x8*)p;
  bf16x8 hi = *(const bf16x8*)(p + 16);
  bf16x16 f;
#pragma unroll
  for (int i = 0; i < 8; ++i) { f[i] = lo[i]; f[i + 8] = hi[i]; }
  return f;
}

DEVFN bf16x16 load_b(const float* B, long ldb, int n0, int k0, int lane) {
  const float* p = B + (long)(n0 + (lane & 15)) * ldb + (k0 + ((lane >> 4) << 3));
  bf16x16 f;
#pragma unroll
  for (int i = 0; i < 4; ++i) {
    f32x4 v = *(const f32x4*)(p + 4 * i + (i >= 2 ? 8 : 0));
#pragma unroll
    for (int e = 0; e < 4; ++e) f[4 * i + e] = (__bf16)v[e];
  }
  return f;
}

DEVFN void store_c(float* p, float v)  { *p = v; }
DEVFN void store_c(__bf16* p, float v) { *p = (__bf16)v; }

// ------------------------------ GEMM (NT) ----------------------------------
// C[M][N] = A[M][K] * B[N][K]^T.  Block = 256 threads = 8 waves (4 M x 2 N),
// wave tile 32x64 -> block tile 128x128.  M,N multiples of 128, K of 32.
// A/B tiles (128x32) staged in LDS by the Tensor Data Mover, double buffered:
// wave 0 issues tensor_load_to_lds for tile k+1, waits TENSORcnt<=2 so tile k
// is resident, all waves barrier then run 8 WMMAs out of LDS.
template <typename AT, typename BT, typename OT>
__global__ void __launch_bounds__(256)
gemm_nt_wmma(const AT* __restrict__ A, long lda,
             const BT* __restrict__ B, long ldb,
             OT* __restrict__ C, long ldc, int K, int Mtot, int Ntot) {
  __shared__ AT shA[2][128 * 32];
  __shared__ BT shB[2][128 * 32];

  const int lane = threadIdx.x & 31;
  const int wave = threadIdx.x >> 5;
  const int mb = blockIdx.x * 128;
  const int nb = blockIdx.y * 128;
  const int m0 = (wave & 3) * 32;     // within block tile
  const int n0 = (wave >> 2) * 64;

  f32x8 acc[2][4];
#pragma unroll
  for (int i = 0; i < 2; ++i)
#pragma unroll
    for (int j = 0; j < 4; ++j) acc[i][j] = f32x8{};

  const unsigned ecA = (sizeof(AT) == 4) ? 2u : 1u;
  const unsigned ecB = (sizeof(BT) == 4) ? 2u : 1u;

  if (wave == 0) {
    tdm_load_2d(lds_addr_of(&shA[0][0]), A + (long)mb * lda,
                (unsigned)lda, (unsigned)Mtot, 32u, 128u, (unsigned)lda, ecA);
    tdm_load_2d(lds_addr_of(&shB[0][0]), B + (long)nb * ldb,
                (unsigned)ldb, (unsigned)Ntot, 32u, 128u, (unsigned)ldb, ecB);
  }

  for (int k0 = 0, it = 0; k0 < K; k0 += 32, ++it) {
    const int cur = it & 1;
    if (wave == 0) {
      if (k0 + 32 < K) {
        tdm_load_2d(lds_addr_of(&shA[cur ^ 1][0]), A + (long)mb * lda + (k0 + 32),
                    (unsigned)lda, (unsigned)Mtot, 32u, 128u, (unsigned)lda, ecA);
        tdm_load_2d(lds_addr_of(&shB[cur ^ 1][0]), B + (long)nb * ldb + (k0 + 32),
                    (unsigned)ldb, (unsigned)Ntot, 32u, 128u, (unsigned)ldb, ecB);
        __builtin_amdgcn_s_wait_tensorcnt(2);   // tile k resident, k+1 in flight
      } else {
        __builtin_amdgcn_s_wait_tensorcnt(0);   // last tile resident
      }
    }
    __syncthreads();                            // publish LDS tile to all waves

    bf16x16 a0 = load_a(&shA[cur][0], 32, m0,      0, lane);
    bf16x16 a1 = load_a(&shA[cur][0], 32, m0 + 16, 0, lane);
    bf16x16 b[4];
#pragma unroll
    for (int j = 0; j < 4; ++j) b[j] = load_b(&shB[cur][0], 32, n0 + 16 * j, 0, lane);
#pragma unroll
    for (int j = 0; j < 4; ++j) {
      acc[0][j] = wmma_bf16(a0, b[j], acc[0][j]);
      acc[1][j] = wmma_bf16(a1, b[j], acc[1][j]);
    }
    __syncthreads();                            // all reads done before buffer reuse
  }

  // C/D layout: lane L -> col (L&15); vgpr r -> row r + 8*(L>=16).
  const int cn    = lane & 15;
  const int rbase = (lane >> 4) << 3;
#pragma unroll
  for (int i = 0; i < 2; ++i)
#pragma unroll
    for (int j = 0; j < 4; ++j)
#pragma unroll
      for (int r = 0; r < 8; ++r) {
        long row = mb + m0 + 16 * i + rbase + r;
        long col = nb + n0 + 16 * j + cn;
        store_c(C + row * ldc + col, acc[i][j][r]);
      }
}

// ------------------------------- RoPE --------------------------------------
__global__ void rope_kernel(__bf16* X, int nheads) {
  int idx = blockIdx.x * blockDim.x + threadIdx.x;
  int total = S_LEN * nheads * (HD / 2);
  if (idx >= total) return;
  int i = idx & 63;           // dim pair 0..63
  int t = idx >> 6;
  int h = t % nheads;
  int s = t / nheads;
  // inv_freq = 10000^(-2i/HD) = exp(-(i/64)*ln(10000))
  float inv = __expf(-(float)i * 0.015625f * 9.210340371976184f);
  float ang = (float)s * inv;
  float c, sn;
  __sincosf(ang, &sn, &c);
  __bf16* p = X + ((long)s * nheads + h) * HD;
  float x1 = (float)p[i];
  float x2 = (float)p[i + 64];
  p[i]      = (__bf16)(x1 * c - x2 * sn);
  p[i + 64] = (__bf16)(x2 * c + x1 * sn);
}

// -------------------- V transpose: [S][NKV*HD] -> [NKV*HD][S] --------------
__global__ void transpose_v(const __bf16* __restrict__ V, __bf16* __restrict__ Vt) {
  int idx = blockIdx.x * blockDim.x + threadIdx.x;
  if (idx >= S_LEN * NKV * HD) return;
  int c = idx & (NKV * HD - 1);
  int s = idx >> 10;
  Vt[(long)c * S_LEN + s] = V[(long)s * (NKV * HD) + c];
}

// --------------------------- Flash attention -------------------------------
// One wave per (head, 16-row q-block). 4 waves / block.
__global__ void __launch_bounds__(128)
attn_wmma(const __bf16* __restrict__ Q, const __bf16* __restrict__ Kc,
          const __bf16* __restrict__ Vt, __bf16* __restrict__ O) {
  const int lane = threadIdx.x & 31;
  const int wave = threadIdx.x >> 5;
  const int head = blockIdx.y;
  const int kvh  = head >> 2;                       // GROUPS = 4
  const int q0   = (blockIdx.x * 4 + wave) * 16;

  __shared__ __bf16 plds[4][16 * 32];               // per-wave P staging
  __bf16* pbuf = &plds[wave][0];

  const long ldq = NH * HD, ldk = NKV * HD;

  bf16x16 aq[4];                                    // Q rows q0..q0+15, HD=128
#pragma unroll
  for (int kk = 0; kk < 4; ++kk)
    aq[kk] = load_a(Q + head * HD, ldq, q0, kk * 32, lane);

  f32x8 oacc[8];
#pragma unroll
  for (int n = 0; n < 8; ++n) oacc[n] = f32x8{};
  float m_run[8], l_run[8];
#pragma unroll
  for (int r = 0; r < 8; ++r) { m_run[r] = -3.0e38f; l_run[r] = 0.0f; }

  const float scale = 0.08838834764831845f;         // 1/sqrt(128)
  const int cn    = lane & 15;
  const int rbase = (lane >> 4) << 3;

  for (int j = 0; j < q0 + 16; j += 32) {           // causal: keys <= q0+15
    f32x8 s0 = f32x8{}, s1 = f32x8{};
#pragma unroll
    for (int kk = 0; kk < 4; ++kk) {
      bf16x16 b0 = load_b(Kc + kvh * HD, ldk, j,      kk * 32, lane);
      bf16x16 b1 = load_b(Kc + kvh * HD, ldk, j + 16, kk * 32, lane);
      s0 = wmma_bf16(aq[kk], b0, s0);
      s1 = wmma_bf16(aq[kk], b1, s1);
    }
    // online softmax; row r's 16 columns live in one 16-lane half
#pragma unroll
    for (int r = 0; r < 8; ++r) {
      const int row = q0 + rbase + r;
      float v0 = s0[r] * scale; if (j + cn > row)      v0 = -1.0e9f;
      float v1 = s1[r] * scale; if (j + 16 + cn > row) v1 = -1.0e9f;
      float mx = fmaxf(v0, v1);
#pragma unroll
      for (int off = 1; off < 16; off <<= 1)
        mx = fmaxf(mx, __shfl_xor(mx, off, 32));
      float mnew = fmaxf(m_run[r], mx);
      float al = __expf(m_run[r] - mnew);
      float p0 = __expf(v0 - mnew);
      float p1 = __expf(v1 - mnew);
      float ps = p0 + p1;
#pragma unroll
      for (int off = 1; off < 16; off <<= 1)
        ps += __shfl_xor(ps, off, 32);
      l_run[r] = l_run[r] * al + ps;
      m_run[r] = mnew;
#pragma unroll
      for (int n = 0; n < 8; ++n) oacc[n][r] *= al;
      const int lr = rbase + r;
      pbuf[lr * 32 + cn]      = (__bf16)p0;
      pbuf[lr * 32 + cn + 16] = (__bf16)p1;
    }
    // P (16x32) back as an A fragment; per-wave LDS, DS ops in-order per wave
    bf16x16 pa = load_a((const __bf16*)pbuf, 32, 0, 0, lane);
#pragma unroll
    for (int n = 0; n < 8; ++n) {
      bf16x16 bv = load_b(Vt + (long)kvh * HD * S_LEN, S_LEN, n * 16, j, lane);
      oacc[n] = wmma_bf16(pa, bv, oacc[n]);
    }
  }

#pragma unroll
  for (int n = 0; n < 8; ++n)
#pragma unroll
    for (int r = 0; r < 8; ++r) {
      long row = q0 + rbase + r;
      O[row * (long)(NH * HD) + head * HD + n * 16 + cn] =
          (__bf16)(oacc[n][r] / l_run[r]);
    }
}

// ------------------------------- launcher ----------------------------------
extern "C" void kernel_launch(void* const* d_in, const int* in_sizes, int n_in,
                              void* d_out, int out_size, void* d_ws, size_t ws_size,
                              hipStream_t stream) {
  const float* X  = (const float*)d_in[0];   // [S][HID]
  const float* Wq = (const float*)d_in[1];   // [NH*HD][HID]
  const float* Wk = (const float*)d_in[2];   // [NKV*HD][HID]
  const float* Wv = (const float*)d_in[3];   // [NKV*HD][HID]
  const float* Wo = (const float*)d_in[4];   // [HID][HID]
  float* out = (float*)d_out;                // [S][HID] fp32

  // workspace layout (44 MB total)
  char* w = (char*)d_ws;
  __bf16* Qb = (__bf16*)(w);                          // S*4096  (16 MB)
  __bf16* Kb = (__bf16*)(w + ((size_t)16 << 20));     // S*1024  ( 4 MB)
  __bf16* Vb = (__bf16*)(w + ((size_t)20 << 20));     // S*1024  ( 4 MB)
  __bf16* Vt = (__bf16*)(w + ((size_t)24 << 20));     // 1024*S  ( 4 MB)
  __bf16* Ob = (__bf16*)(w + ((size_t)28 << 20));     // S*4096  (16 MB)

  dim3 blk(256);
  // QKV projections (fp32 in, bf16 out, f32 accumulate in WMMA)
  gemm_nt_wmma<float, float, __bf16>
      <<<dim3(S_LEN / 128, HID / 128), blk, 0, stream>>>(
          X, HID, Wq, HID, Qb, HID, HID, S_LEN, HID);
  gemm_nt_wmma<float, float, __bf16>
      <<<dim3(S_LEN / 128, (NKV * HD) / 128), blk, 0, stream>>>(
          X, HID, Wk, HID, Kb, NKV * HD, HID, S_LEN, NKV * HD);
  gemm_nt_wmma<float, float, __bf16>
      <<<dim3(S_LEN / 128, (NKV * HD) / 128), blk, 0, stream>>>(
          X, HID, Wv, HID, Vb, NKV * HD, HID, S_LEN, NKV * HD);

  rope_kernel<<<(S_LEN * NH  * (HD / 2)) / 256, 256, 0, stream>>>(Qb, NH);
  rope_kernel<<<(S_LEN * NKV * (HD / 2)) / 256, 256, 0, stream>>>(Kb, NKV);

  transpose_v<<<(S_LEN * NKV * HD) / 256, 256, 0, stream>>>(Vb, Vt);

  attn_wmma<<<dim3(S_LEN / 16 / 4, NH), dim3(128), 0, stream>>>(Qb, Kb, Vt, Ob);

  // output projection (bf16 x fp32 -> fp32 out)
  gemm_nt_wmma<__bf16, float, float>
      <<<dim3(S_LEN / 128, HID / 128), blk, 0, stream>>>(
          Ob, HID, Wo, HID, out, HID, HID, S_LEN, HID);
}
